// SpaceTimeTransformer_55104430407910
// MI455X (gfx1250) — compile-verified, matrix operand order checked by reading, not run on hardware
//
#include <hip/hip_runtime.h>
#include <hip/hip_bf16.h>

typedef __attribute__((ext_vector_type(16))) __bf16 v16bf;
typedef __attribute__((ext_vector_type(8)))  __bf16 v8bf;
typedef __attribute__((ext_vector_type(8)))  float  v8f;

// ---- problem constants (from reference setup_inputs) ----
#define DD    768      // model dim
#define NTOK  1569     // 1 + F*n
#define BATCH 8
#define HEADS 12
#define HD    64       // head dim
#define FR    8        // frames
#define SP    196      // spatial tokens per frame
#define BH    (BATCH*HEADS)   // 96
#define TT    (BATCH*NTOK)    // 12552 tokens total
#define NPAD  1584     // padded per-(bh) token rows
#define KPAD  224      // padded key count for space attn (= 14*16 = 7*32)
#define KT    14       // key tiles

// ---------------- bf16 helpers ----------------
__device__ __forceinline__ float bf2f(__bf16 b) {
  unsigned short h = __builtin_bit_cast(unsigned short, b);
  unsigned u = ((unsigned)h) << 16;
  return __builtin_bit_cast(float, u);
}
__device__ __forceinline__ __bf16 f2bf(float f) {
  unsigned u = __builtin_bit_cast(unsigned, f);
  u += 0x7FFFu + ((u >> 16) & 1u);             // round-to-nearest-even
  unsigned short h = (unsigned short)(u >> 16);
  return __builtin_bit_cast(__bf16, h);
}

// Load one 16-bit WMMA A/B fragment for this lane.
// p points at element k0 = kstep + ((lane&16)?8:0); fragment elements
// 0..7 = p[0..7], 8..15 = p[16..23]  (two 16B-aligned b128 loads).
__device__ __forceinline__ v16bf ldfrag(const __bf16* p) {
  const v8bf* q = (const v8bf*)p;
  v8bf lo = q[0];
  v8bf hi = q[2];
  return __builtin_shufflevector(lo, hi, 0,1,2,3,4,5,6,7,8,9,10,11,12,13,14,15);
}

#define WMMA_BF16(A_, B_, C_) \
  __builtin_amdgcn_wmma_f32_16x16x32_bf16(false, (A_), false, (B_), (short)0, (C_), false, false)

// ---- CDNA5 async global->LDS copy (ASYNCcnt-tracked) ----
__device__ __forceinline__ void async_copy_b128(void* lds, const void* g) {
  asm volatile("global_load_async_to_lds_b128 %0, %1, off"
               :: "v"((unsigned)(uintptr_t)lds),          // LDS byte address (addr[31:0])
                  "v"((unsigned long long)(uintptr_t)g)   // 64-bit global address
               : "memory");
}
__device__ __forceinline__ void wait_async0() {
  asm volatile("s_wait_asynccnt 0" ::: "memory");
}

// ---------------- fp32 (K x N) -> bf16 transposed (N x K) ----------------
__global__ void k_cvt_bf16_t(const float* __restrict__ in, __bf16* __restrict__ out,
                             int K, int N) {
  long long total = (long long)K * N;
  long long i = (long long)blockIdx.x * blockDim.x + threadIdx.x;
  long long st = (long long)gridDim.x * blockDim.x;
  for (; i < total; i += st) {
    int k = (int)(i / N), n = (int)(i % N);
    out[(size_t)n * K + k] = f2bf(in[i]);
  }
}

// ---------------- LayerNorm (row of 768), fp32 in, bf16 out ----------------
__global__ void k_layernorm(const float* __restrict__ x, const float* __restrict__ g,
                            const float* __restrict__ b, __bf16* __restrict__ out) {
  int row = blockIdx.x;
  int tid = threadIdx.x;
  const float* xr = x + (size_t)row * DD;
  __shared__ float red[256];
  float v0 = xr[tid], v1 = xr[tid + 256], v2 = xr[tid + 512];
  red[tid] = v0 + v1 + v2;
  __syncthreads();
  for (int o = 128; o > 0; o >>= 1) { if (tid < o) red[tid] += red[tid + o]; __syncthreads(); }
  float mean = red[0] * (1.0f / DD);
  __syncthreads();
  float d0 = v0 - mean, d1 = v1 - mean, d2 = v2 - mean;
  red[tid] = d0 * d0 + d1 * d1 + d2 * d2;
  __syncthreads();
  for (int o = 128; o > 0; o >>= 1) { if (tid < o) red[tid] += red[tid + o]; __syncthreads(); }
  float inv = rsqrtf(red[0] * (1.0f / DD) + 1e-5f);
  __bf16* orow = out + (size_t)row * DD;
  orow[tid]       = f2bf(d0 * inv * g[tid]       + b[tid]);
  orow[tid + 256] = f2bf(d1 * inv * g[tid + 256] + b[tid + 256]);
  orow[tid + 512] = f2bf(d2 * inv * g[tid + 512] + b[tid + 512]);
}

// ---------------- generic WMMA GEMM (2x4 tiles per wave) ----------------
// C = A(MxK,bf16,row-major) @ W^T(Wt is N x K, bf16) + bias
// 256 threads = 8 waves; wave w computes row tiles (blockIdx.y*16 + 2w .. +1), 4 col tiles.
// mode 0: outf = res + result (fp32)   mode 1: outb = bf16(result)
// mode 2: outb = bf16(gelu_exact(result))
__global__ void k_gemm(const __bf16* __restrict__ A, const __bf16* __restrict__ Wt,
                       const float* __restrict__ bias, const float* __restrict__ res,
                       float* __restrict__ outf, __bf16* __restrict__ outb,
                       int M, int N, int K, int mode) {
  int tid = threadIdx.x;
  int wave = tid >> 5, lane = tid & 31;
  int tm = blockIdx.y * 256 + wave * 32;       // two row tiles: tm, tm+16
  if (tm >= M) return;                          // wave-uniform
  int tn = blockIdx.x * 64;
  int r0 = tm + (lane & 15);      if (r0 >= M) r0 = M - 1;   // clamp; bad rows discarded at store
  int r1 = tm + 16 + (lane & 15); if (r1 >= M) r1 = M - 1;
  int k0 = (lane & 16) ? 8 : 0;
  const __bf16* Ap0 = A  + (size_t)r0 * K + k0;
  const __bf16* Ap1 = A  + (size_t)r1 * K + k0;
  const __bf16* Wp  = Wt + (size_t)(tn + (lane & 15)) * K + k0;
  v8f c00 = {}, c01 = {}, c02 = {}, c03 = {};
  v8f c10 = {}, c11 = {}, c12 = {}, c13 = {};
  for (int kk = 0; kk < K; kk += 32) {
    if (kk + 64 < K) {
      __builtin_prefetch(Ap0 + kk + 64, 0, 1);
      __builtin_prefetch(Ap1 + kk + 64, 0, 1);
    }
    v16bf a0 = ldfrag(Ap0 + kk);
    v16bf a1 = ldfrag(Ap1 + kk);
    v16bf b0 = ldfrag(Wp + kk);
    v16bf b1 = ldfrag(Wp + (size_t)16 * K + kk);
    v16bf b2 = ldfrag(Wp + (size_t)32 * K + kk);
    v16bf b3 = ldfrag(Wp + (size_t)48 * K + kk);
    c00 = WMMA_BF16(a0, b0, c00);  c10 = WMMA_BF16(a1, b0, c10);
    c01 = WMMA_BF16(a0, b1, c01);  c11 = WMMA_BF16(a1, b1, c11);
    c02 = WMMA_BF16(a0, b2, c02);  c12 = WMMA_BF16(a1, b2, c12);
    c03 = WMMA_BF16(a0, b3, c03);  c13 = WMMA_BF16(a1, b3, c13);
  }
  v8f accs[2][4] = {{c00, c01, c02, c03}, {c10, c11, c12, c13}};
  #pragma unroll
  for (int rt = 0; rt < 2; rt++) {
    #pragma unroll
    for (int t = 0; t < 4; t++) {
      int col = tn + t * 16 + (lane & 15);
      float bv = bias[col];
      #pragma unroll
      for (int r = 0; r < 8; r++) {
        int row = tm + rt * 16 + r + ((lane & 16) ? 8 : 0);
        if (row >= M) continue;
        float v = accs[rt][t][r] + bv;
        size_t idx = (size_t)row * N + col;
        if (mode == 0) {
          outf[idx] = (res ? res[idx] : 0.0f) + v;
        } else if (mode == 1) {
          outb[idx] = f2bf(v);
        } else {
          float gv = 0.5f * v * (1.0f + erff(v * 0.70710678118654752f)); // exact GELU
          outb[idx] = f2bf(gv);
        }
      }
    }
  }
}

// ---------------- QKV GEMM: scatter to head-major Q/K/V (bf16), Q pre-scaled ----------------
__global__ void k_gemm_qkv(const __bf16* __restrict__ A, const __bf16* __restrict__ Wt,
                           const float* __restrict__ bias,
                           __bf16* __restrict__ Q, __bf16* __restrict__ Kb, __bf16* __restrict__ V) {
  const int M = TT, K = DD;
  int tid = threadIdx.x;
  int wave = tid >> 5, lane = tid & 31;
  int tm = blockIdx.y * 256 + wave * 32;
  if (tm >= M) return;
  int tn = blockIdx.x * 64;
  int r0 = tm + (lane & 15);      if (r0 >= M) r0 = M - 1;
  int r1 = tm + 16 + (lane & 15); if (r1 >= M) r1 = M - 1;
  int k0 = (lane & 16) ? 8 : 0;
  const __bf16* Ap0 = A  + (size_t)r0 * K + k0;
  const __bf16* Ap1 = A  + (size_t)r1 * K + k0;
  const __bf16* Wp  = Wt + (size_t)(tn + (lane & 15)) * K + k0;
  v8f c00 = {}, c01 = {}, c02 = {}, c03 = {};
  v8f c10 = {}, c11 = {}, c12 = {}, c13 = {};
  for (int kk = 0; kk < K; kk += 32) {
    if (kk + 64 < K) {
      __builtin_prefetch(Ap0 + kk + 64, 0, 1);
      __builtin_prefetch(Ap1 + kk + 64, 0, 1);
    }
    v16bf a0 = ldfrag(Ap0 + kk);
    v16bf a1 = ldfrag(Ap1 + kk);
    v16bf b0 = ldfrag(Wp + kk);
    v16bf b1 = ldfrag(Wp + (size_t)16 * K + kk);
    v16bf b2 = ldfrag(Wp + (size_t)32 * K + kk);
    v16bf b3 = ldfrag(Wp + (size_t)48 * K + kk);
    c00 = WMMA_BF16(a0, b0, c00);  c10 = WMMA_BF16(a1, b0, c10);
    c01 = WMMA_BF16(a0, b1, c01);  c11 = WMMA_BF16(a1, b1, c11);
    c02 = WMMA_BF16(a0, b2, c02);  c12 = WMMA_BF16(a1, b2, c12);
    c03 = WMMA_BF16(a0, b3, c03);  c13 = WMMA_BF16(a1, b3, c13);
  }
  v8f accs[2][4] = {{c00, c01, c02, c03}, {c10, c11, c12, c13}};
  #pragma unroll
  for (int rt = 0; rt < 2; rt++) {
    #pragma unroll
    for (int t = 0; t < 4; t++) {
      int col = tn + t * 16 + (lane & 15);
      int which = col / DD;          // 0=q 1=k 2=v  (64-col tile never crosses a 768 boundary)
      int hc = col % DD;
      int head = hc >> 6, d = hc & 63;
      float bv = bias[col];
      #pragma unroll
      for (int r = 0; r < 8; r++) {
        int row = tm + rt * 16 + r + ((lane & 16) ? 8 : 0);
        if (row >= M) continue;
        int b = row / NTOK, pos = row % NTOK;
        size_t idx = (((size_t)(b * HEADS + head)) * NPAD + pos) * HD + d;
        float v = accs[rt][t][r] + bv;
        if (which == 0)      Q[idx]  = f2bf(v * 0.125f);   // hd^-0.5
        else if (which == 1) Kb[idx] = f2bf(v);
        else                 V[idx]  = f2bf(v);
      }
    }
  }
}

// ---------------- CLS attention: 1 query over all 1569 keys, per (b,h) ----------------
__global__ void k_cls_attn(const __bf16* __restrict__ Qg, const __bf16* __restrict__ Kg,
                           const __bf16* __restrict__ Vg, __bf16* __restrict__ attn) {
  int bh = blockIdx.x, tid = threadIdx.x;
  int b = bh / HEADS, h = bh % HEADS;
  size_t base = (size_t)bh * NPAD * HD;
  __shared__ float qv[HD];
  __shared__ float lg[NTOK];
  __shared__ float red[256];
  __shared__ float osum[4][HD];
  if (tid < HD) qv[tid] = bf2f(Qg[base + tid]);
  __syncthreads();
  float lmax = -3.4e38f;
  for (int j = tid; j < NTOK; j += 256) {
    const __bf16* kr = Kg + base + (size_t)j * HD;
    float s = 0.0f;
    for (int d = 0; d < HD; d++) s += qv[d] * bf2f(kr[d]);
    lg[j] = s;
    lmax = fmaxf(lmax, s);
  }
  red[tid] = lmax; __syncthreads();
  for (int o = 128; o > 0; o >>= 1) { if (tid < o) red[tid] = fmaxf(red[tid], red[tid + o]); __syncthreads(); }
  float bmax = red[0]; __syncthreads();
  float lsum = 0.0f;
  for (int j = tid; j < NTOK; j += 256) { float e = __expf(lg[j] - bmax); lg[j] = e; lsum += e; }
  red[tid] = lsum; __syncthreads();
  for (int o = 128; o > 0; o >>= 1) { if (tid < o) red[tid] += red[tid + o]; __syncthreads(); }
  float inv = 1.0f / red[0];
  __syncthreads();
  int g = tid >> 6, d = tid & 63;
  float o = 0.0f;
  for (int j = g; j < NTOK; j += 4) o += lg[j] * bf2f(Vg[base + (size_t)j * HD + d]);
  osum[g][d] = o;
  __syncthreads();
  if (g == 0) {
    float t = (osum[0][d] + osum[1][d] + osum[2][d] + osum[3][d]) * inv;
    attn[((size_t)b * NTOK) * DD + h * HD + d] = f2bf(t);
  }
}

// ---------------- time attention: per (bh, spatial pos): 8 queries x 9 keys ----------------
__global__ void k_time_attn(const __bf16* __restrict__ Qg, const __bf16* __restrict__ Kg,
                            const __bf16* __restrict__ Vg, __bf16* __restrict__ attn) {
  int bh = blockIdx.x / SP, ni = blockIdx.x % SP;
  int b = bh / HEADS, h = bh % HEADS;
  size_t base = (size_t)bh * NPAD * HD;
  __shared__ float Ql[FR][HD], Kl[9][HD], Vl[9][HD], P[FR][9];
  int tid = threadIdx.x;
  for (int idx = tid; idx < 9 * HD; idx += 128) {
    int j = idx / HD, d = idx % HD;
    int pos = (j == 0) ? 0 : (1 + (j - 1) * SP + ni);
    size_t a = base + (size_t)pos * HD + d;
    Kl[j][d] = bf2f(Kg[a]);
    Vl[j][d] = bf2f(Vg[a]);
  }
  for (int idx = tid; idx < FR * HD; idx += 128) {
    int qi = idx / HD, d = idx % HD;
    int pos = 1 + qi * SP + ni;
    Ql[qi][d] = bf2f(Qg[base + (size_t)pos * HD + d]);
  }
  __syncthreads();
  if (tid < FR * 9) {
    int qi = tid / 9, kj = tid % 9;
    float s = 0.0f;
    for (int d = 0; d < HD; d++) s += Ql[qi][d] * Kl[kj][d];
    P[qi][kj] = s;
  }
  __syncthreads();
  if (tid < FR) {
    float m = -3.4e38f;
    for (int j = 0; j < 9; j++) m = fmaxf(m, P[tid][j]);
    float s = 0.0f;
    for (int j = 0; j < 9; j++) { float e = __expf(P[tid][j] - m); P[tid][j] = e; s += e; }
    float inv = 1.0f / s;
    for (int j = 0; j < 9; j++) P[tid][j] *= inv;
  }
  __syncthreads();
  for (int idx = tid; idx < FR * HD; idx += 128) {
    int qi = idx >> 6, d = idx & 63;
    float o = 0.0f;
    for (int j = 0; j < 9; j++) o += P[qi][j] * Vl[j][d];
    int pos = 1 + qi * SP + ni;
    attn[((size_t)b * NTOK + pos) * DD + h * HD + d] = f2bf(o);
  }
}

// ---------------- space attention (WMMA): per (bh, frame): 196 queries x 197 keys ----------------
// 4 waves / block. K rows + V rows async-copied global->LDS (ASYNCcnt), V then
// transposed in LDS; QK^T by WMMA; in-register softmax (shfl_xor across 16-lane
// halves); probs -> LDS; P@V by WMMA.
__global__ void k_space_attn(const __bf16* __restrict__ Qg, const __bf16* __restrict__ Kg,
                             const __bf16* __restrict__ Vg, __bf16* __restrict__ attn) {
  int bh = blockIdx.x >> 3, frame = blockIdx.x & 7;
  int b = bh / HEADS, h = bh % HEADS;
  size_t base = (size_t)bh * NPAD * HD;
  extern __shared__ __bf16 smem[];
  __bf16* Kl = smem;                       // [KPAD][HD]
  __bf16* Vt = Kl + KPAD * HD;             // [HD][KPAD]   (transposed V)
  __bf16* Pb = Vt + KPAD * HD;             // 4 waves * [16][KPAD]; doubles as V-row temp
  int tid = threadIdx.x;
  int wave = tid >> 5, lane = tid & 31;
  __bf16* Vrows = Pb;                      // KPAD*HD == 4*16*KPAD elements (HD==64)
  // ---- stage K rows and V rows via CDNA5 async copy; zero K padding
  for (int idx = tid; idx < KPAD * (HD / 8); idx += 128) {
    int j = idx / (HD / 8);
    int db = (idx % (HD / 8)) * 8;
    if (j < 1 + SP) {
      int pos = (j == 0) ? 0 : (1 + frame * SP + (j - 1));
      size_t a = base + (size_t)pos * HD + db;
      async_copy_b128(Kl + j * HD + db, Kg + a);
      async_copy_b128(Vrows + j * HD + db, Vg + a);
    } else {
      v8bf zv = {};
      *(v8bf*)(Kl + j * HD + db) = zv;
    }
  }
  wait_async0();
  __syncthreads();
  // ---- transpose V rows -> Vt; zero key padding columns
  for (int idx = tid; idx < KPAD * (HD / 8); idx += 128) {
    int j = idx / (HD / 8);
    int db = (idx % (HD / 8)) * 8;
    if (j < 1 + SP) {
      v8bf vv = *(const v8bf*)(Vrows + j * HD + db);
      #pragma unroll
      for (int e = 0; e < 8; e++) Vt[(db + e) * KPAD + j] = vv[e];
    } else {
      __bf16 z = f2bf(0.0f);
      #pragma unroll
      for (int e = 0; e < 8; e++) Vt[(db + e) * KPAD + j] = z;
    }
  }
  __syncthreads();
  __bf16* Pw = Pb + wave * 16 * KPAD;
  int qposbase = 1 + frame * SP;
  int k0 = (lane & 16) ? 8 : 0;
  for (int rt = wave; rt < 13; rt += 4) {          // 13 row tiles cover 196 rows
    v8f zz = {};
    v8f accs[KT];
    #pragma unroll
    for (int kt = 0; kt < KT; kt++) accs[kt] = zz;
    // ---- logits = Q_tile @ K^T
    const __bf16* Qp = Qg + base + (size_t)(qposbase + rt * 16 + (lane & 15)) * HD + k0;
    #pragma unroll
    for (int kk = 0; kk < HD; kk += 32) {
      v16bf af = ldfrag(Qp + kk);
      #pragma unroll
      for (int kt = 0; kt < KT; kt++) {
        v16bf bfr = ldfrag(Kl + (kt * 16 + (lane & 15)) * HD + kk + k0);
        accs[kt] = WMMA_BF16(af, bfr, accs[kt]);
      }
    }
    // ---- mask padded keys
    #pragma unroll
    for (int kt = 0; kt < KT; kt++) {
      int j = kt * 16 + (lane & 15);
      if (j >= 1 + SP) {
        #pragma unroll
        for (int r = 0; r < 8; r++) accs[kt][r] = -1.0e30f;
      }
    }
    // ---- softmax over keys (rows live across 16-lane halves)
    float rm[8], rs[8];
    #pragma unroll
    for (int r = 0; r < 8; r++) {
      float m = accs[0][r];
      #pragma unroll
      for (int kt = 1; kt < KT; kt++) m = fmaxf(m, accs[kt][r]);
      #pragma unroll
      for (int msk = 1; msk < 16; msk <<= 1) m = fmaxf(m, __shfl_xor(m, msk));
      rm[r] = m;
    }
    #pragma unroll
    for (int r = 0; r < 8; r++) {
      float s = 0.0f;
      #pragma unroll
      for (int kt = 0; kt < KT; kt++) { float e = __expf(accs[kt][r] - rm[r]); accs[kt][r] = e; s += e; }
      #pragma unroll
      for (int msk = 1; msk < 16; msk <<= 1) s += __shfl_xor(s, msk);
      rs[r] = 1.0f / s;
    }
    // ---- probs (bf16) to this wave's LDS region (C-layout -> row-major)
    #pragma unroll
    for (int kt = 0; kt < KT; kt++) {
      int j = kt * 16 + (lane & 15);
      #pragma unroll
      for (int r = 0; r < 8; r++) {
        int row = r + ((lane & 16) ? 8 : 0);
        Pw[row * KPAD + j] = f2bf(accs[kt][r] * rs[r]);
      }
    }
    asm volatile("s_wait_dscnt 0" ::: "memory");   // wave-local LDS RAW fence
    // ---- out = P @ V
    #pragma unroll
    for (int nt = 0; nt < 4; nt++) {
      v8f acc = zz;
      const __bf16* Pp = Pw + (lane & 15) * KPAD + k0;
      const __bf16* Vp = Vt + (size_t)(nt * 16 + (lane & 15)) * KPAD + k0;
      #pragma unroll
      for (int kk = 0; kk < KPAD; kk += 32) {
        v16bf af  = ldfrag(Pp + kk);
        v16bf bfr = ldfrag(Vp + kk);
        acc = WMMA_BF16(af, bfr, acc);
      }
      #pragma unroll
      for (int r = 0; r < 8; r++) {
        int lr = rt * 16 + r + ((lane & 16) ? 8 : 0);
        if (lr < SP) {
          int pos = qposbase + lr;
          attn[((size_t)b * NTOK + pos) * DD + h * HD + nt * 16 + (lane & 15)] = f2bf(acc[r]);
        }
      }
    }
  }
}

// ================================================================
extern "C" void kernel_launch(void* const* d_in, const int* in_sizes, int n_in,
                              void* d_out, int out_size, void* d_ws, size_t ws_size,
                              hipStream_t stream) {
  (void)in_sizes; (void)n_in; (void)out_size; (void)ws_size;
  const float* x      = (const float*)d_in[0];
  const float* n1g    = (const float*)d_in[1];
  const float* n1b    = (const float*)d_in[2];
  const float* n2g    = (const float*)d_in[3];
  const float* n2b    = (const float*)d_in[4];
  const float* n3g    = (const float*)d_in[5];
  const float* n3b    = (const float*)d_in[6];
  const float* aqkvw  = (const float*)d_in[7];
  const float* aqkvb  = (const float*)d_in[8];
  const float* aprojw = (const float*)d_in[9];
  const float* aprojb = (const float*)d_in[10];
  const float* tqkvw  = (const float*)d_in[11];
  const float* tqkvb  = (const float*)d_in[12];
  const float* tprojw = (const float*)d_in[13];
  const float* tprojb = (const float*)d_in[14];
  const float* fc1w   = (const float*)d_in[15];
  const float* fc1b   = (const float*)d_in[16];
  const float* fc2w   = (const float*)d_in[17];
  const float* fc2b   = (const float*)d_in[18];
  float* out = (float*)d_out;

  char* ws = (char*)d_ws;
  size_t off = 0;
  auto alloc = [&](size_t bytes) -> char* {
    char* p = ws + off;
    off += (bytes + 255) & ~(size_t)255;
    return p;
  };
  // transposed bf16 weights: Wt is (N x K)
  __bf16* w_aqkv = (__bf16*)alloc((size_t)DD * 3 * DD * 2);
  __bf16* w_tqkv = (__bf16*)alloc((size_t)DD * 3 * DD * 2);
  __bf16* w_aprj = (__bf16*)alloc((size_t)DD * DD * 2);
  __bf16* w_tprj = (__bf16*)alloc((size_t)DD * DD * 2);
  __bf16* w_fc1  = (__bf16*)alloc((size_t)DD * 4 * DD * 2);
  __bf16* w_fc2  = (__bf16*)alloc((size_t)4 * DD * DD * 2);
  __bf16* lnb    = (__bf16*)alloc((size_t)TT * DD * 2);
  __bf16* Qb     = (__bf16*)alloc((size_t)BH * NPAD * HD * 2);
  __bf16* Kb     = (__bf16*)alloc((size_t)BH * NPAD * HD * 2);
  __bf16* Vb     = (__bf16*)alloc((size_t)BH * NPAD * HD * 2);
  __bf16* attnb  = (__bf16*)alloc((size_t)TT * DD * 2);
  float*  TR     = (float*) alloc((size_t)TT * DD * 4);
  float*  SR     = (float*) alloc((size_t)TT * DD * 4);
  __bf16* Hb     = (__bf16*)alloc((size_t)TT * 4 * DD * 2);

  // weights -> bf16 transposed (every call; deterministic)
  k_cvt_bf16_t<<<256, 256, 0, stream>>>(aqkvw,  w_aqkv, DD, 3 * DD);
  k_cvt_bf16_t<<<256, 256, 0, stream>>>(tqkvw,  w_tqkv, DD, 3 * DD);
  k_cvt_bf16_t<<<256, 256, 0, stream>>>(aprojw, w_aprj, DD, DD);
  k_cvt_bf16_t<<<256, 256, 0, stream>>>(tprojw, w_tprj, DD, DD);
  k_cvt_bf16_t<<<256, 256, 0, stream>>>(fc1w,   w_fc1,  DD, 4 * DD);
  k_cvt_bf16_t<<<256, 256, 0, stream>>>(fc2w,   w_fc2,  4 * DD, DD);

  dim3 blk(256);
  int mblocks = (TT + 255) / 256;               // 2 row-tiles per wave, 8 waves per block
  dim3 gQKV((3 * DD) / 64, mblocks);
  dim3 gPRJ(DD / 64,       mblocks);
  dim3 gFC1((4 * DD) / 64, mblocks);
  dim3 gFC2(DD / 64,       mblocks);
  const size_t space_lds = (size_t)(2 * KPAD * HD + 4 * 16 * KPAD) * sizeof(__bf16); // 86016 B

  // ---- time branch: x + attn_time(LN3(x))
  k_layernorm<<<TT, 256, 0, stream>>>(x, n3g, n3b, lnb);
  k_gemm_qkv<<<gQKV, blk, 0, stream>>>(lnb, w_tqkv, tqkvb, Qb, Kb, Vb);
  k_cls_attn<<<BH, 256, 0, stream>>>(Qb, Kb, Vb, attnb);
  k_time_attn<<<BH * SP, 128, 0, stream>>>(Qb, Kb, Vb, attnb);
  k_gemm<<<gPRJ, blk, 0, stream>>>(attnb, w_tprj, tprojb, x, TR, nullptr, TT, DD, DD, 0);

  // ---- space branch: x + attn_space(LN1(time_residual))
  k_layernorm<<<TT, 256, 0, stream>>>(TR, n1g, n1b, lnb);
  k_gemm_qkv<<<gQKV, blk, 0, stream>>>(lnb, w_aqkv, aqkvb, Qb, Kb, Vb);
  k_cls_attn<<<BH, 256, 0, stream>>>(Qb, Kb, Vb, attnb);
  k_space_attn<<<BH * FR, 128, space_lds, stream>>>(Qb, Kb, Vb, attnb);
  k_gemm<<<gPRJ, blk, 0, stream>>>(attnb, w_aprj, aprojb, x, SR, nullptr, TT, DD, DD, 0);

  // ---- MLP: SR + fc2(gelu(fc1(LN2(SR))))
  k_layernorm<<<TT, 256, 0, stream>>>(SR, n2g, n2b, lnb);
  k_gemm<<<gFC1, blk, 0, stream>>>(lnb, w_fc1, fc1b, nullptr, nullptr, Hb, TT, 4 * DD, DD, 2);
  k_gemm<<<gFC2, blk, 0, stream>>>(Hb, w_fc2, fc2b, SR, out, nullptr, TT, DD, 4 * DD, 0);
}
